// RPContrastiveLoss_48601849922124
// MI455X (gfx1250) — matrix operand order repaired and checked by reading, not compiled, not run on hardware
//
#include <hip/hip_runtime.h>
#include <math.h>

#define ROI 5
#define NUM_CLASSES 80
#define MEM_SIZE 1024

typedef __attribute__((ext_vector_type(16))) __bf16 v16bf;
typedef __attribute__((ext_vector_type(8)))  __bf16 v8bf;
typedef __attribute__((ext_vector_type(8)))  float  v8f;

// ---------------------------------------------------------------------------
// Stage 1: ROI align.  One block per (roi, sample-point); 256 threads = chans.
// ---------------------------------------------------------------------------
__global__ __launch_bounds__(256) void roi_align_kernel(
    const float* __restrict__ rp, const int* __restrict__ bidx,
    const float* __restrict__ bboxes, float* __restrict__ feats,
    int C, int H, int W) {
  int blk = blockIdx.x;
  int n = blk / (ROI * ROI);
  int p = blk % (ROI * ROI);
  int iy = p / ROI, ix = p % ROI;

  float cx = bboxes[n * 4 + 0], cy = bboxes[n * 4 + 1];
  float bw = bboxes[n * 4 + 2], bh = bboxes[n * 4 + 3];
  float x1 = cx - bw * 0.5f, y1 = cy - bh * 0.5f;
  float x2 = cx + bw * 0.5f, y2 = cy + bh * 0.5f;
  float roi_w = fmaxf(x2 - x1, 1.0f), roi_h = fmaxf(y2 - y1, 1.0f);
  float x = x1 + ((float)ix + 0.5f) * (roi_w / (float)ROI);
  float y = y1 + ((float)iy + 0.5f) * (roi_h / (float)ROI);
  bool invalid = (y < -1.0f) || (y > (float)H) || (x < -1.0f) || (x > (float)W);
  y = fmaxf(y, 0.0f); x = fmaxf(x, 0.0f);
  int yl = (int)floorf(y), xl = (int)floorf(x);
  if (yl >= H - 1) y = (float)(H - 1);
  if (xl >= W - 1) x = (float)(W - 1);
  yl = min(yl, H - 1); xl = min(xl, W - 1);
  int yh = min(yl + 1, H - 1), xh = min(xl + 1, W - 1);
  float ly = y - (float)yl, lx = x - (float)xl;
  float hy = 1.0f - ly, hx = 1.0f - lx;

  int b = bidx[n];
  const float* base = rp + (size_t)b * C * H * W;
  for (int c = threadIdx.x; c < C; c += blockDim.x) {
    const float* pc = base + (size_t)c * H * W;
    float v = pc[yl * W + xl] * hy * hx + pc[yl * W + xh] * hy * lx +
              pc[yh * W + xl] * ly * hx + pc[yh * W + xh] * ly * lx;
    if (invalid) v = 0.0f;
    feats[(size_t)n * (C * ROI * ROI) + (size_t)c * (ROI * ROI) + p] = v;
  }
}

// ---------------------------------------------------------------------------
// Stage 2/3: Out[M,N] = act(A[M,K] @ Wt[N,K]^T + bias).  bf16 WMMA, f32 acc.
// Tile: 64M x 128N x 32K.  256 threads = 8 waves, each wave a 32x32 subtile.
// ---------------------------------------------------------------------------
template <bool RELU>
__global__ __launch_bounds__(256) void gemm_bf16wmma_kernel(
    const float* __restrict__ A, const float* __restrict__ Wt,
    const float* __restrict__ bias, float* __restrict__ Out,
    int M, int N, int K, float eps) {
  constexpr int TM = 64, TN = 128, TK = 32, PAD = 8;
  __shared__ __bf16 lA[TM][TK + PAD];
  __shared__ __bf16 lB[TN][TK + PAD];

  const int tid = threadIdx.x;
  const int mBase = blockIdx.x * TM;
  const int nBase = blockIdx.y * TN;
  const int lane = tid & 31;
  const int wave = tid >> 5;
  const int wm = (wave >> 2) * 32;        // 0 or 32
  const int wn = (wave & 3) * 32;         // 0,32,64,96
  const int mrow = lane & 15;
  const int k0a = (lane < 16) ? 0 : 8;    // A frag: K{k0a..k0a+7, k0a+16..k0a+23}
  const int k0b = (lane < 16) ? 0 : 8;    // B frag: K{k0b..k0b+7, k0b+16..k0b+23}

  v8f acc[2][2];
#pragma unroll
  for (int mt = 0; mt < 2; ++mt)
#pragma unroll
    for (int nt = 0; nt < 2; ++nt)
#pragma unroll
      for (int r = 0; r < 8; ++r) acc[mt][nt][r] = 0.0f;

  for (int kb = 0; kb < K; kb += TK) {
    // stage A tile: 64x32 f32 -> bf16 (512 float4, 2 per thread)
#pragma unroll
    for (int q = 0; q < 2; ++q) {
      int e = q * 256 + tid;
      int row = e >> 3;
      int kc = (e & 7) * 4;
      const float4 v = *(const float4*)(A + (size_t)(mBase + row) * K + kb + kc);
      lA[row][kc + 0] = (__bf16)v.x; lA[row][kc + 1] = (__bf16)v.y;
      lA[row][kc + 2] = (__bf16)v.z; lA[row][kc + 3] = (__bf16)v.w;
    }
    // stage B tile: 128x32 f32 -> bf16 (1024 float4, 4 per thread)
#pragma unroll
    for (int q = 0; q < 4; ++q) {
      int e = q * 256 + tid;
      int row = e >> 3;
      int kc = (e & 7) * 4;
      const float4 v = *(const float4*)(Wt + (size_t)(nBase + row) * K + kb + kc);
      lB[row][kc + 0] = (__bf16)v.x; lB[row][kc + 1] = (__bf16)v.y;
      lB[row][kc + 2] = (__bf16)v.z; lB[row][kc + 3] = (__bf16)v.w;
    }
    __syncthreads();

    v16bf af[2], bfr[2];
#pragma unroll
    for (int mt = 0; mt < 2; ++mt) {
      const v8bf lo = *(const v8bf*)&lA[wm + mt * 16 + mrow][k0a];
      const v8bf hi = *(const v8bf*)&lA[wm + mt * 16 + mrow][k0a + 16];
#pragma unroll
      for (int i = 0; i < 8; ++i) { af[mt][i] = lo[i]; af[mt][8 + i] = hi[i]; }
    }
#pragma unroll
    for (int nt = 0; nt < 2; ++nt) {
      const v8bf lo = *(const v8bf*)&lB[wn + nt * 16 + mrow][k0b];
      const v8bf hi = *(const v8bf*)&lB[wn + nt * 16 + mrow][k0b + 16];
#pragma unroll
      for (int i = 0; i < 8; ++i) { bfr[nt][i] = lo[i]; bfr[nt][8 + i] = hi[i]; }
    }
#pragma unroll
    for (int mt = 0; mt < 2; ++mt)
#pragma unroll
      for (int nt = 0; nt < 2; ++nt)
        acc[mt][nt] = __builtin_amdgcn_wmma_f32_16x16x32_bf16(
            false, af[mt], false, bfr[nt], (short)0, acc[mt][nt], false, false);
    __syncthreads();
  }

  // epilogue: D layout — VGPR r: lanes<16 = row r, lanes>=16 = row 8+r; col = lane&15
#pragma unroll
  for (int mt = 0; mt < 2; ++mt)
#pragma unroll
    for (int nt = 0; nt < 2; ++nt) {
      int col = nBase + wn + nt * 16 + (lane & 15);
      float bv = bias[col];
#pragma unroll
      for (int r = 0; r < 8; ++r) {
        int row = mBase + wm + mt * 16 + ((lane < 16) ? r : 8 + r);
        float v = acc[mt][nt][r] + bv;
        if (RELU) v = fmaxf(v, 0.0f); else v += eps;
        Out[(size_t)row * N + col] = v;
      }
    }
}

// ---------------------------------------------------------------------------
// Stage 3b: row-wise L2 normalization (512 rows x 256)
// ---------------------------------------------------------------------------
__global__ __launch_bounds__(256) void normalize_kernel(
    const float* __restrict__ f, float* __restrict__ out, int D) {
  __shared__ float red[256];
  int i = blockIdx.x, t = threadIdx.x;
  float v = f[(size_t)i * D + t];
  red[t] = v * v;
  __syncthreads();
  for (int s = 128; s > 0; s >>= 1) {
    if (t < s) red[t] += red[t + s];
    __syncthreads();
  }
  float den = fmaxf(sqrtf(red[0]), 1e-12f);
  out[(size_t)i * D + t] = v / den;
}

// ---------------------------------------------------------------------------
// Stage 4a: bulk copy memory bank (84 MB) -- float4 grid-stride + prefetch
// ---------------------------------------------------------------------------
__global__ __launch_bounds__(256) void copy4_kernel(
    const float4* __restrict__ src, float4* __restrict__ dst, size_t n) {
  size_t i = (size_t)blockIdx.x * blockDim.x + threadIdx.x;
  size_t stride = (size_t)gridDim.x * blockDim.x;
  for (; i < n; i += stride) {
    if (i + stride < n) __builtin_prefetch(src + i + stride, 0, 0);
    dst[i] = src[i];
  }
}

// ---------------------------------------------------------------------------
// Stage 4b: occurrence positions, counts, class weights. Single block.
// ---------------------------------------------------------------------------
__global__ __launch_bounds__(512) void stats_kernel(
    const int* __restrict__ cls, const int* __restrict__ mcount,
    const float* __restrict__ cfreq, int* __restrict__ pos,
    float* __restrict__ out_count, float* __restrict__ out_cw, int N) {
  __shared__ int scls[512];
  __shared__ int perCls[NUM_CLASSES];
  __shared__ float cwRaw[NUM_CLASSES];
  __shared__ float sTotal, sMean;
  int t = threadIdx.x;
  if (t < NUM_CLASSES) perCls[t] = 0;
  scls[t] = cls[t];
  __syncthreads();

  int c = scls[t];
  int occ = 0;
  for (int j = 0; j < t; ++j) occ += (scls[j] == c) ? 1 : 0;
  pos[t] = (mcount[c] + occ) % MEM_SIZE;
  atomicAdd(&perCls[c], 1);
  __syncthreads();

  if (t == 0) {
    float tot = 0.0f;
    for (int k = 0; k < NUM_CLASSES; ++k) tot += cfreq[k] + (float)perCls[k];
    sTotal = tot;
  }
  __syncthreads();

  float cwv = 0.0f;
  if (t < NUM_CLASSES) {
    out_count[t] = (float)(mcount[t] + perCls[t]);
    float fq = cfreq[t] + (float)perCls[t];
    float prob = fmaxf(fq / sTotal, 1e-8f);
    cwv = 1.0f / sqrtf(prob + 1e-8f);
    cwRaw[t] = cwv;
  }
  __syncthreads();

  if (t == 0) {
    float s = 0.0f;
    for (int k = 0; k < NUM_CLASSES; ++k) s += cwRaw[k];
    sMean = s / (float)NUM_CLASSES;
  }
  __syncthreads();
  if (t < NUM_CLASSES) out_cw[t] = cwv / sMean;
}

// ---------------------------------------------------------------------------
// Stage 4c: scatter normalized features into copied memory bank
// ---------------------------------------------------------------------------
__global__ __launch_bounds__(256) void scatter_kernel(
    const float* __restrict__ featn, const int* __restrict__ cls,
    const int* __restrict__ pos, float* __restrict__ mb, int D) {
  int i = blockIdx.x, t = threadIdx.x;
  int c = cls[i], p = pos[i];
  mb[((size_t)c * MEM_SIZE + p) * D + t] = featn[(size_t)i * D + t];
}

// ---------------------------------------------------------------------------
extern "C" void kernel_launch(void* const* d_in, const int* in_sizes, int n_in,
                              void* d_out, int out_size, void* d_ws, size_t ws_size,
                              hipStream_t stream) {
  const float* rp     = (const float*)d_in[0];   // (16,256,128,128)
  const int*   bidx   = (const int*)d_in[1];     // (512,)
  const int*   cls    = (const int*)d_in[2];     // (512,)
  const float* bboxes = (const float*)d_in[3];   // (512,4)
  // d_in[4], d_in[5] = img_h, img_w (unused by reference math)
  const float* w1     = (const float*)d_in[6];   // (2048,6400)
  const float* b1     = (const float*)d_in[7];   // (2048,)
  const float* w2     = (const float*)d_in[8];   // (256,2048)
  const float* b2     = (const float*)d_in[9];   // (256,)
  const float* mbank  = (const float*)d_in[10];  // (80,1024,256)
  const int*   mcount = (const int*)d_in[11];    // (80,)
  const float* cfreq  = (const float*)d_in[12];  // (80,)

  const int C = 256, H = 128, W = 128, N = 512;
  const int ENC = C * ROI * ROI;  // 6400
  const int HID = 2048, DIM = 256;

  // workspace layout
  float* feats = (float*)d_ws;                         // 512*6400
  float* hbuf  = feats + (size_t)N * ENC;              // 512*2048
  float* fbuf  = hbuf + (size_t)N * HID;               // 512*256
  int*   pos   = (int*)(fbuf + (size_t)N * DIM);       // 512

  // output layout: feat_norm | memory_bank | count_new | cw
  float* out_feat = (float*)d_out;
  float* out_mb   = out_feat + (size_t)N * DIM;
  float* out_cnt  = out_mb + (size_t)NUM_CLASSES * MEM_SIZE * DIM;
  float* out_cw   = out_cnt + NUM_CLASSES;

  roi_align_kernel<<<N * ROI * ROI, 256, 0, stream>>>(rp, bidx, bboxes, feats, C, H, W);

  dim3 g1(N / 64, HID / 128);
  gemm_bf16wmma_kernel<true><<<g1, 256, 0, stream>>>(feats, w1, b1, hbuf, N, HID, ENC, 0.0f);

  dim3 g2(N / 64, DIM / 128);
  gemm_bf16wmma_kernel<false><<<g2, 256, 0, stream>>>(hbuf, w2, b2, fbuf, N, DIM, HID, 1e-8f);

  normalize_kernel<<<N, DIM, 0, stream>>>(fbuf, out_feat, DIM);

  size_t mbQuads = (size_t)NUM_CLASSES * MEM_SIZE * DIM / 4;
  copy4_kernel<<<2048, 256, 0, stream>>>((const float4*)mbank, (float4*)out_mb, mbQuads);

  stats_kernel<<<1, 512, 0, stream>>>(cls, mcount, cfreq, pos, out_cnt, out_cw, N);

  scatter_kernel<<<N, DIM, 0, stream>>>(out_feat, cls, pos, out_mb, DIM);
}
